// TFCorrelationDomainClassifier_64321430225217
// MI455X (gfx1250) — matrix operand order repaired and checked by reading, not compiled, not run on hardware
//
#include <hip/hip_runtime.h>
#include <hip/hip_bf16.h>

typedef _Float16 half_t;
typedef __attribute__((ext_vector_type(16))) _Float16 v16h;
typedef __attribute__((ext_vector_type(8)))  _Float16 v8h;
typedef __attribute__((ext_vector_type(8)))  float    v8f;

#define BDIM 2048
#define FDIM 512
#define CDIM 64

union HFrag { v16h v; v8h h[2]; };

__device__ inline v8f wmma_f16(v16h a, v16h b, v8f c) {
  // D = A(16x32 f16) * B(32x16 f16) + C(16x16 f32)
  return __builtin_amdgcn_wmma_f32_16x16x32_f16(
      /*neg_a=*/false, a, /*neg_b=*/false, b,
      /*c_mod=*/(short)0, c, /*reuse_a=*/false, /*reuse_b=*/false);
}

// ---------------------------------------------------------------- converts
__global__ void cvt_f32_to_f16(const float* __restrict__ in,
                               half_t* __restrict__ out, int n) {
  int i = blockIdx.x * blockDim.x + threadIdx.x;
  int stride = gridDim.x * blockDim.x;
  for (; i < n; i += stride) out[i] = (half_t)in[i];
}

// W_proj[c, i*F + j] (fp32)  ->  WT[c, j*F + i] (f16), tiled transpose via LDS
__global__ void w_transpose_f16(const float* __restrict__ W,
                                half_t* __restrict__ WT) {
  __shared__ half_t tile[32][33];
  const int c  = blockIdx.z;
  const int i0 = blockIdx.x * 32;
  const int j0 = blockIdx.y * 32;
  const float*  Wc  = W  + (size_t)c * FDIM * FDIM;
  half_t*       WTc = WT + (size_t)c * FDIM * FDIM;
  #pragma unroll
  for (int rr = 0; rr < 32; rr += 8) {
    int i = i0 + threadIdx.y + rr;
    int j = j0 + threadIdx.x;
    tile[threadIdx.y + rr][threadIdx.x] = (half_t)Wc[(size_t)i * FDIM + j];
  }
  __syncthreads();
  #pragma unroll
  for (int rr = 0; rr < 32; rr += 8) {
    int j = j0 + threadIdx.y + rr;
    int i = i0 + threadIdx.x;
    WTc[(size_t)j * FDIM + i] = tile[threadIdx.x][threadIdx.y + rr];
  }
}

// ---------------------------------------------------------------- main fused bilinear
// grid = (B/32, C), block = 256 (8 waves). Each block: 2 batch tiles of 16 x one c.
// Wave w owns columns j in [w*64, w*64+64).
__global__ __launch_bounds__(256)
void bilinear_kernel(const half_t* __restrict__ t_h,
                     const half_t* __restrict__ f_h,
                     const half_t* __restrict__ WT,  // [c][j][i], f16
                     const float*  __restrict__ b_proj,
                     float* __restrict__ proj) {
  __shared__ half_t Msh[2 * 8 * 16 * 64];  // [bt][wave][row*64+col]  32 KB
  __shared__ float  Psh[8 * 2 * 16 * 16];  // [wave][bt][row*16+col]  16 KB

  const int c    = blockIdx.y;
  const int b0   = blockIdx.x * 32;
  const int wave = threadIdx.x >> 5;
  const int lane = threadIdx.x & 31;
  const int r    = lane & 15;       // fragment row/col index
  const int half = lane >> 4;       // lane-half selects K sub-range
  const int jbase = wave * 64;

  const half_t* Wc = WT + (size_t)c * FDIM * FDIM;

  // ---------------- Stage 1: M[bt][16 x 64] = T_tile(16xF) * Wr[c](FxF) slice
  v8f acc[2][4];
  #pragma unroll
  for (int bt = 0; bt < 2; ++bt)
    #pragma unroll
    for (int nt = 0; nt < 4; ++nt)
      acc[bt][nt] = (v8f){0.f, 0.f, 0.f, 0.f, 0.f, 0.f, 0.f, 0.f};

  #pragma unroll 4
  for (int k0 = 0; k0 < FDIM; k0 += 32) {
    HFrag a[2];
    #pragma unroll
    for (int bt = 0; bt < 2; ++bt) {
      const half_t* arow = t_h + (size_t)(b0 + bt * 16 + r) * FDIM + k0;
      a[bt].h[0] = *(const v8h*)(arow + half * 8);        // K = k0 + half*8 + 0..7
      a[bt].h[1] = *(const v8h*)(arow + 16 + half * 8);   // K = k0+16+half*8 + 0..7
    }
    #pragma unroll
    for (int nt = 0; nt < 4; ++nt) {
      const int j = jbase + nt * 16 + r;                  // B column index
      const half_t* bcol = Wc + (size_t)j * FDIM + k0 + half * 8;
      HFrag b;
      b.h[0] = *(const v8h*)(bcol);                       // K = k0 + half*8 + 0..7
      b.h[1] = *(const v8h*)(bcol + 16);                  // K = k0+16+half*8 + 0..7
      acc[0][nt] = wmma_f16(a[0].v, b.v, acc[0][nt]);
      acc[1][nt] = wmma_f16(a[1].v, b.v, acc[1][nt]);
    }
  }

  // Spill M to LDS as f16 (C-layout -> row-major 16x64 per wave, per bt)
  #pragma unroll
  for (int bt = 0; bt < 2; ++bt) {
    half_t* m = &Msh[(bt * 8 + wave) * 1024];
    #pragma unroll
    for (int nt = 0; nt < 4; ++nt)
      #pragma unroll
      for (int v = 0; v < 8; ++v)
        m[(v + 8 * half) * 64 + nt * 16 + r] = (half_t)acc[bt][nt][v];
  }
  __syncthreads();

  // ---------------- Stage 2: P_partial[bt] = M_chunk(16x64) * F_chunkT(64x16)
  v8f p[2];
  p[0] = (v8f){0.f, 0.f, 0.f, 0.f, 0.f, 0.f, 0.f, 0.f};
  p[1] = (v8f){0.f, 0.f, 0.f, 0.f, 0.f, 0.f, 0.f, 0.f};
  #pragma unroll
  for (int t = 0; t < 2; ++t) {  // two K=32 steps over this wave's 64 columns
    #pragma unroll
    for (int bt = 0; bt < 2; ++bt) {
      const half_t* mrow = &Msh[(bt * 8 + wave) * 1024 + r * 64 + t * 32];
      HFrag a;
      a.h[0] = *(const v8h*)(mrow + half * 8);
      a.h[1] = *(const v8h*)(mrow + 16 + half * 8);
      // B[k=j_local][n=row] = f_h[b0+bt*16+n, jbase + t*32 + k]; column contiguous
      const half_t* fcol = f_h + (size_t)(b0 + bt * 16 + r) * FDIM
                         + jbase + t * 32 + half * 8;
      HFrag b;
      b.h[0] = *(const v8h*)(fcol);
      b.h[1] = *(const v8h*)(fcol + 16);
      p[bt] = wmma_f16(a.v, b.v, p[bt]);
    }
  }

  // Write 16x16 partials, reduce over waves, keep the diagonal
  #pragma unroll
  for (int bt = 0; bt < 2; ++bt) {
    float* ps = &Psh[(wave * 2 + bt) * 256];
    #pragma unroll
    for (int v = 0; v < 8; ++v)
      ps[(v + 8 * half) * 16 + r] = p[bt][v];
  }
  __syncthreads();

  if (threadIdx.x < 32) {
    const int bt = threadIdx.x >> 4;
    const int rr = threadIdx.x & 15;
    float s = 0.f;
    #pragma unroll
    for (int w = 0; w < 8; ++w) s += Psh[(w * 2 + bt) * 256 + rr * 16 + rr];
    s += b_proj[c];
    proj[(size_t)(b0 + bt * 16 + rr) * CDIM + c] = s;
  }
}

// ---------------------------------------------------------------- MLP head
__global__ void mlp_head(const float* __restrict__ proj,
                         const float* __restrict__ W1, const float* __restrict__ b1,
                         const float* __restrict__ W2, const float* __restrict__ b2,
                         float* __restrict__ out) {
  int b = blockIdx.x * blockDim.x + threadIdx.x;
  if (b >= BDIM) return;
  float h[CDIM];
  #pragma unroll
  for (int k = 0; k < CDIM; ++k) {
    float v = proj[(size_t)b * CDIM + k];
    h[k] = v > 0.f ? v : 0.f;                    // ReLU (GRL = identity fwd)
  }
  float acc2 = b2[0];
  for (int o = 0; o < 32; ++o) {
    float s = b1[o];
    #pragma unroll
    for (int k = 0; k < CDIM; ++k) s += W1[o * CDIM + k] * h[k];
    s = s > 0.f ? s : 0.f;
    acc2 += W2[o] * s;
  }
  out[b] = acc2;
}

// ---------------------------------------------------------------- launch
extern "C" void kernel_launch(void* const* d_in, const int* in_sizes, int n_in,
                              void* d_out, int out_size, void* d_ws, size_t ws_size,
                              hipStream_t stream) {
  const float* t_feat = (const float*)d_in[0];
  const float* f_feat = (const float*)d_in[1];
  const float* W_proj = (const float*)d_in[2];
  const float* b_proj = (const float*)d_in[3];
  const float* W1     = (const float*)d_in[4];
  const float* b1     = (const float*)d_in[5];
  const float* W2     = (const float*)d_in[6];
  const float* b2     = (const float*)d_in[7];

  char* ws = (char*)d_ws;
  half_t* t_h  = (half_t*)(ws);                                  // 2 MB
  half_t* f_h  = (half_t*)(ws + (size_t)(2 << 20));              // 2 MB
  half_t* WT   = (half_t*)(ws + (size_t)(4 << 20));              // 32 MB
  float*  proj = (float*) (ws + (size_t)(36 << 20));             // 512 KB

  cvt_f32_to_f16<<<1024, 256, 0, stream>>>(t_feat, t_h, BDIM * FDIM);
  cvt_f32_to_f16<<<1024, 256, 0, stream>>>(f_feat, f_h, BDIM * FDIM);

  dim3 gT(FDIM / 32, FDIM / 32, CDIM);
  w_transpose_f16<<<gT, dim3(32, 8), 0, stream>>>(W_proj, WT);

  dim3 gM(BDIM / 32, CDIM);
  bilinear_kernel<<<gM, 256, 0, stream>>>(t_h, f_h, WT, b_proj, proj);

  mlp_head<<<(BDIM + 255) / 256, 256, 0, stream>>>(proj, W1, b1, W2, b2,
                                                   (float*)d_out);
}
